// C_Encoder_78434692760248
// MI455X (gfx1250) — matrix operand order repaired and checked by reading, not compiled, not run on hardware
//
#include <hip/hip_runtime.h>

typedef __attribute__((ext_vector_type(16))) _Float16 v16h;
typedef __attribute__((ext_vector_type(8)))  _Float16 v8h;
typedef __attribute__((ext_vector_type(4)))  _Float16 v4h;
typedef __attribute__((ext_vector_type(8)))  float    v8f;

#define DD 256
#define HS 264          // padded LDS row stride in halfs (528 B -> 4-bank rotation)
#define WAVES 4
#define ROWS_PER_WAVE 16
#define ROWS_PER_WG (WAVES * ROWS_PER_WAVE)

union V16 { v16h v; v8h h[2]; };

__device__ __forceinline__ float tanh_fast(float x) {
    // tanh(x) = 1 - 2/(e^{2x}+1); saturates correctly for |x| large (exp->inf/0)
    float e = __expf(2.0f * x);
    return 1.0f - 2.0f * __builtin_amdgcn_rcpf(e + 1.0f);
}

// ---- LN stats over one 256-wide LDS row; lane (rw,hh) sums its half, halves
// combined with shfl_xor(16). Returns mu, rsigma. ----
__device__ __forceinline__ void ln_stats(const _Float16* __restrict__ myh,
                                         int rw, int hh, float& mu, float& rs) {
    float s = 0.f, ss = 0.f;
    const _Float16* hrow = myh + rw * HS + hh * 128;
    #pragma unroll
    for (int j = 0; j < 16; ++j) {
        v8h d = *(const v8h*)(hrow + j * 8);
        #pragma unroll
        for (int i = 0; i < 8; ++i) { float f = (float)d[i]; s += f; ss += f * f; }
    }
    s  += __shfl_xor(s, 16, 32);
    ss += __shfl_xor(ss, 16, 32);
    mu        = s  * (1.0f / 256.0f);
    float var = ss * (1.0f / 256.0f) - mu * mu;
    rs        = rsqrtf(var + 1e-5f);
}

// ---- build normalized A fragments (ISA 16-bit A 16x32 layout):
//  lane half 0: K = kf*32 + [0..8) and [16..24); half 1: +8 shift ----
__device__ __forceinline__ void build_A(v16h* __restrict__ A,
                                        const _Float16* __restrict__ myh,
                                        int rw, int hh, float mu, float rs) {
    _Float16 muh = (_Float16)mu, rsh = (_Float16)rs;
    v8h mus = {muh, muh, muh, muh, muh, muh, muh, muh};
    v8h rss = {rsh, rsh, rsh, rsh, rsh, rsh, rsh, rsh};
    #pragma unroll
    for (int kf = 0; kf < 8; ++kf) {
        int base = kf * 32 + hh * 8;
        v8h lo = *(const v8h*)(myh + rw * HS + base);
        v8h hv = *(const v8h*)(myh + rw * HS + base + 16);
        lo = (lo - mus) * rss;
        hv = (hv - mus) * rss;
        V16 u; u.h[0] = lo; u.h[1] = hv;
        A[kf] = u.v;
    }
}

// ---- one 16x16 output tile: 8 K-step WMMAs; B fragments straight from the
// L2-resident f16 weights (lane = N, K halfs [8h,8h+8) and [16+8h,24+8h) -> two b128) ----
__device__ __forceinline__ v8f gemm_tile(const v16h* __restrict__ A,
                                         const _Float16* __restrict__ wp) {
    v8f c = {};
    #pragma unroll
    for (int kf = 0; kf < 8; ++kf) {
        V16 ub;
        ub.h[0] = *(const v8h*)(wp + kf * 32);
        ub.h[1] = *(const v8h*)(wp + kf * 32 + 16);
        c = __builtin_amdgcn_wmma_f32_16x16x32_f16(
                false, A[kf], false, ub.v, (short)0, c, false, false);
    }
    return c;
}

// Fold LN affine into weights/bias and convert weights to f16:
//   Wh[L][o][k] = f16(W[o,k] * g[k])
//   bias[L][o]  = b_lin[o] + sum_k W[o,k] * b_ln[k]
__global__ __launch_bounds__(256) void prep_kernel(
    const float* __restrict__ W1, const float* __restrict__ b1,
    const float* __restrict__ g1, const float* __restrict__ n1,
    const float* __restrict__ W2, const float* __restrict__ b2,
    const float* __restrict__ g2, const float* __restrict__ n2,
    const float* __restrict__ W3, const float* __restrict__ b3,
    const float* __restrict__ g3, const float* __restrict__ n3,
    _Float16* __restrict__ Wh, float* __restrict__ bias)
{
    int L = blockIdx.x >> 8;     // 0..5
    int o = blockIdx.x & 255;    // output row
    int k = threadIdx.x;         // input col
    const float *W, *g, *bln, *blin;
    if (L < 4)       { W = W1 + L * 65536; g = g1 + L * 256; bln = n1 + L * 256; blin = b1 + L * 256; }
    else if (L == 4) { W = W2;             g = g2;           bln = n2;           blin = b2; }
    else             { W = W3;             g = g3;           bln = n3;           blin = b3; }

    float wv = W[o * 256 + k];
    Wh[(size_t)L * 65536 + o * 256 + k] = (_Float16)(wv * g[k]);

    __shared__ float red[256];
    red[k] = wv * bln[k];
    __syncthreads();
    #pragma unroll
    for (int s = 128; s > 0; s >>= 1) {
        if (k < s) red[k] += red[k + s];
        __syncthreads();
    }
    if (k == 0) bias[L * 256 + o] = blin[o] + red[0];
}

__global__ __launch_bounds__(WAVES * 32) void encoder_kernel(
    const float* __restrict__ x, const _Float16* __restrict__ Wh,
    const float* __restrict__ bias, float* __restrict__ out)
{
    __shared__ _Float16 hsm[WAVES][ROWS_PER_WAVE][HS];   // 33792 B, per-wave private slices

    const int tid = threadIdx.x;
    const int w   = tid >> 5;     // wave in workgroup
    const int l   = tid & 31;     // lane
    const int rw  = l & 15;       // A-layout row (M) / B,C-layout column (N) within tile
    const int hh  = l >> 4;       // lane half (selects K sub-range per ISA layouts)
    const int r0  = (blockIdx.x * WAVES + w) * ROWS_PER_WAVE;

    _Float16* myh = &hsm[w][0][0];

    // ---- stage x tile (f32 global -> f16 LDS), coalesced 512B per row-load ----
    for (int r = 0; r < ROWS_PER_WAVE; ++r) {
        const float4* xr = (const float4*)(x + (size_t)(r0 + r) * DD);
        float4 fa = xr[l];
        float4 fb = xr[l + 32];
        v4h ha, hb;
        ha[0] = (_Float16)fa.x; ha[1] = (_Float16)fa.y; ha[2] = (_Float16)fa.z; ha[3] = (_Float16)fa.w;
        hb[0] = (_Float16)fb.x; hb[1] = (_Float16)fb.y; hb[2] = (_Float16)fb.z; hb[3] = (_Float16)fb.w;
        *(v4h*)(myh + r * HS + l * 4)       = ha;
        *(v4h*)(myh + r * HS + 128 + l * 4) = hb;
    }

    v16h A[8];
    float mu, rs;

    // ================= 4 hidden layers: LN -> GEMM -> tanh, h updated in place =================
    for (int L = 0; L < 4; ++L) {
        ln_stats(myh, rw, hh, mu, rs);
        build_A(A, myh, rw, hh, mu, rs);

        const _Float16* WL = Wh + (size_t)L * 65536;
        const float*    bL = bias + L * 256;

        for (int nt = 0; nt < 16; ++nt) {
            int n = nt * 16 + rw;                        // output column (lane = N)
            v8f c = gemm_tile(A, WL + n * 256 + hh * 8);
            float bn = bL[n];
            #pragma unroll
            for (int v = 0; v < 8; ++v) {                // C layout: M = v + 8*hh, N = rw
                float y = tanh_fast(c[v] + bn);
                myh[(v + 8 * hh) * HS + nt * 16 + rw] = (_Float16)y;
            }
        }
    }

    // ================= final mean/var branches: shared LN stats + A fragments =================
    ln_stats(myh, rw, hh, mu, rs);
    build_A(A, myh, rw, hh, mu, rs);

    #pragma unroll
    for (int br = 0; br < 2; ++br) {
        const _Float16* WL = Wh + (size_t)(4 + br) * 65536;
        const float*    bL = bias + (4 + br) * 256;

        for (int nt = 0; nt < 16; ++nt) {
            int n = nt * 16 + rw;
            v8f c = gemm_tile(A, WL + n * 256 + hh * 8);
            float bn = bL[n];
            // output: [mean | var] concat along dim 1, row stride 512
            float* orow = out + (size_t)(r0 + 8 * hh) * 512 + (size_t)br * 256
                              + nt * 16 + rw;
            #pragma unroll
            for (int v = 0; v < 8; ++v) {
                orow[(size_t)v * 512] = c[v] + bn;
            }
        }
    }
}

extern "C" void kernel_launch(void* const* d_in, const int* in_sizes, int n_in,
                              void* d_out, int out_size, void* d_ws, size_t ws_size,
                              hipStream_t stream)
{
    const float* x  = (const float*)d_in[0];
    const float* g1 = (const float*)d_in[1];
    const float* n1 = (const float*)d_in[2];
    const float* W1 = (const float*)d_in[3];
    const float* b1 = (const float*)d_in[4];
    const float* g2 = (const float*)d_in[5];
    const float* n2 = (const float*)d_in[6];
    const float* W2 = (const float*)d_in[7];
    const float* b2 = (const float*)d_in[8];
    const float* g3 = (const float*)d_in[9];
    const float* n3 = (const float*)d_in[10];
    const float* W3 = (const float*)d_in[11];
    const float* b3 = (const float*)d_in[12];
    float* out = (float*)d_out;

    // workspace: 6 layers of f16 weights (768 KB) + 6x256 f32 fused biases
    _Float16* Wh  = (_Float16*)d_ws;
    float*    bia = (float*)((char*)d_ws + (size_t)6 * 65536 * sizeof(_Float16));

    prep_kernel<<<6 * 256, 256, 0, stream>>>(W1, b1, g1, n1,
                                             W2, b2, g2, n2,
                                             W3, b3, g3, n3, Wh, bia);

    const int N = in_sizes[0] / DD;   // 262144 rows
    encoder_kernel<<<N / ROWS_PER_WG, WAVES * 32, 0, stream>>>(x, Wh, bia, out);

    (void)n_in; (void)out_size; (void)ws_size;
}
